// GATSocialEncoder_3212635538175
// MI455X (gfx1250) — matrix operand order repaired and checked by reading, not compiled, not run on hardware
//
#include <hip/hip_runtime.h>
#include <hip/hip_bf16.h>
#include <cstdint>
#include <cstddef>

// ---------------------------------------------------------------------------
// CDNA5 (gfx1250) fused GAT social encoder.
//   Encoder GEMMs run on v_wmma_f32_16x16x32_f16 (wave32 WMMA).
//   4 sequences (T=4) packed per 16-row WMMA tile; 1 block (4 waves) = 1 target
//   batch element b, covering its 32 neighbors; GAT phase fused in-block (LDS).
//   Neighbor histories staged to LDS with GLOBAL_LOAD_ASYNC_TO_LDS_B128
//   (ASYNCcnt) so the encoder front-end reads LDS only.
// ---------------------------------------------------------------------------

typedef __attribute__((ext_vector_type(16))) _Float16 v16h;
typedef __attribute__((ext_vector_type(8)))  float    v8f;

__device__ __forceinline__ v8f wmma16(v16h a, v16h b, v8f c) {
  // 8 args: (neg_a, A, neg_b, B, c_mod, C, reuse_a, reuse_b)
  return __builtin_amdgcn_wmma_f32_16x16x32_f16(false, a, false, b, (short)0, c,
                                                false, false);
}

// A-operand (16x32 f16) gather from row-major f32 [row][ld] storage.
// ISA layout: lane = m + 16*((k>>3)&1); element e -> k = (e>>3)*16 + kh*8 + (e&7)
__device__ __forceinline__ v16h load_a_f32(const float* X, int row, int ld,
                                           int kh, int k0) {
  v16h r;
  const float* p = X + row * ld + k0 + kh * 8;
#pragma unroll
  for (int j = 0; j < 8; ++j) {
    const int k = ((j >> 2) * 16) + ((j & 3) << 1);
    r[2 * j]     = (_Float16)p[k];
    r[2 * j + 1] = (_Float16)p[k + 1];
  }
  return r;
}

// A-operand gather from row-major f16 [row][ld] storage.
__device__ __forceinline__ v16h load_a_f16(const _Float16* X, int row, int ld,
                                           int kh, int k0) {
  v16h r;
  const _Float16* p = X + row * ld + k0 + kh * 8;
#pragma unroll
  for (int j = 0; j < 8; ++j) {
    const int k = ((j >> 2) * 16) + ((j & 3) << 1);
    r[2 * j]     = p[k];
    r[2 * j + 1] = p[k + 1];
  }
  return r;
}

// B-operand (32x16 f16): element e -> k = (e>>3)*16 + kh*8 + (e&7).
// For x @ W.T with W row-major [n][k], this is two 16-byte chunks per lane.
__device__ __forceinline__ v16h load_b_f16(const _Float16* W, int n, int ld,
                                           int kh, int k0) {
  const _Float16* p = W + n * ld + k0 + kh * 8;
  union { v16h h; uint4 u[2]; } un;
  un.u[0] = ((const uint4*)p)[0];
  un.u[1] = ((const uint4*)(p + 16))[0];
  return un.h;
}

// LayerNorm over a 64-wide row of f32 LDS; two lanes (kh=0/1) cooperate per row
// (each recomputes stats, writes its 32-column half). Intra-wave LDS is
// in-order, and all reads precede all writes in the instruction stream.
__device__ __forceinline__ void layernorm_rowhalf(float* X, int row, int kh,
                                                  const float* g,
                                                  const float* be) {
  float m = 0.f;
#pragma unroll
  for (int c = 0; c < 64; ++c) m += X[row * 64 + c];
  m *= (1.f / 64.f);
  float va = 0.f;
#pragma unroll
  for (int c = 0; c < 64; ++c) {
    const float d = X[row * 64 + c] - m;
    va += d * d;
  }
  va *= (1.f / 64.f);
  const float inv = rsqrtf(va + 1e-5f);
#pragma unroll
  for (int cc = 0; cc < 32; ++cc) {
    const int c = kh * 32 + cc;
    X[row * 64 + c] = (X[row * 64 + c] - m) * inv * g[c] + be[c];
  }
}

// ---------------------------------------------------------------------------
// Weight prep: f32 -> f16, packed into d_ws.
//   layer l at halves [l*32768): qkv(12288) out(4096) ff1(8192) ff2(8192)
//   node_w zero-padded [64][96] at halves [65536).  Total 71680 halves.
// ---------------------------------------------------------------------------
struct PrepArgs {
  const float* qkv_w[2];
  const float* out_w[2];
  const float* ff1_w[2];
  const float* ff2_w[2];
  const float* node_w;
};

__global__ void prep_weights_f16(PrepArgs A, _Float16* w16) {
  const int tid = blockIdx.x * blockDim.x + threadIdx.x;
  const int stride = gridDim.x * blockDim.x;
#pragma unroll 1
  for (int l = 0; l < 2; ++l) {
    _Float16* base = w16 + l * 32768;
    for (int i = tid; i < 12288; i += stride) base[i]         = (_Float16)A.qkv_w[l][i];
    for (int i = tid; i < 4096;  i += stride) base[12288 + i] = (_Float16)A.out_w[l][i];
    for (int i = tid; i < 8192;  i += stride) base[16384 + i] = (_Float16)A.ff1_w[l][i];
    for (int i = tid; i < 8192;  i += stride) base[24576 + i] = (_Float16)A.ff2_w[l][i];
  }
  for (int i = tid; i < 64 * 96; i += stride) {
    const int r = i / 96, c = i % 96;
    w16[65536 + i] = (c < 72) ? (_Float16)A.node_w[r * 72 + c] : (_Float16)0.0f;
  }
}

// ---------------------------------------------------------------------------
// Fused main kernel: 1 block (128 threads / 4 wave32) per batch element.
// ---------------------------------------------------------------------------
struct MainArgs {
  const float* nb_hists;     // [B][32][4][4]
  const int*   nb_types;     // [B][32]
  const int*   nb_mask;      // [B][32]
  const float* target_hist;  // [B][4][4]
  const float* in_w;         // [64][4]
  const float* in_b;         // [64]
  const float* pe;           // [20][64] (rows 0..3 used)
  const float* type_emb;     // [2][8]
  const float* qkv_b[2];
  const float* out_b[2];
  const float* ln1_g[2];
  const float* ln1_b[2];
  const float* ff1_b[2];
  const float* ff2_b[2];
  const float* ln2_g[2];
  const float* ln2_b[2];
  const float* edge_w;       // [4][4]
  const float* a_src;        // [16]
  const float* a_dst;        // [16]
  const float* gout_w;       // [64][64]
  const float* gout_b;       // [64]
  const _Float16* w16;       // packed f16 weights
  float* out;                // [B][64]
};

__launch_bounds__(128)
__global__ void gat_social_fused(MainArgs P) {
  // Per-wave scratch (4 waves):
  __shared__ __align__(16) float    xbuf[4][16 * 64];    // activations (f32)
  __shared__ __align__(16) _Float16 sbuf[4][16 * 192];   // qkv / attn-out / ff (f16)
  // Block-level buffers:
  __shared__ __align__(16) float    histlds[32 * 16];    // staged nb_hists (2KB)
  __shared__ __align__(16) _Float16 nbembed[32 * 96];    // [n][72] zero-padded to 96
  __shared__ __align__(16) float    hbuf[32 * 64];       // node projections
  __shared__ float logitb[32 * 4];
  __shared__ float wattn[32 * 4];
  __shared__ float wsum[64];

  const int b    = blockIdx.x;
  const int tid  = threadIdx.x;
  const int lane = tid & 31;
  const int wv   = tid >> 5;   // wave 0..3
  const int l16  = lane & 15;
  const int kh   = lane >> 4;  // lane-half selector for WMMA operands

  float*    X = xbuf[wv];      // [16][64]
  _Float16* S = sbuf[wv];      // [16][192]

  // ---- stage this block's neighbor histories into LDS asynchronously ------
  // 128 threads x 16B = 2KB, tracked by ASYNCcnt. Wave w's loads cover exactly
  // the neighbors its own tiles consume (tid/32 == neighbor/8), so only the
  // per-wave s_wait_asynccnt is needed before the encoder; cross-wave readers
  // (edge features) sit behind later __syncthreads.
  {
    const float* gsrc = P.nb_hists + (size_t)b * 512 + tid * 4;
    const unsigned lds_dst = (unsigned)(size_t)(const void*)(histlds + tid * 4);
    asm volatile("global_load_async_to_lds_b128 %0, %1, off"
                 :: "v"(lds_dst), "v"(gsrc)
                 : "memory");
  }
  // Warm L2/WGP$ with the first weight block (layer-0 QKV region).
  __builtin_prefetch((const void*)(P.w16 + (size_t)tid * 96), 0, 0);

  asm volatile("s_wait_asynccnt 0x0" ::: "memory");

  // ======================= temporal transformer encoder ====================
#pragma unroll 1
  for (int ti = 0; ti < 2; ++ti) {
    const int tile = wv * 2 + ti;  // 8 tiles -> 32 neighbors
    const int nb0  = tile * 4;

    // ---- input projection (4->64) + positional encoding -------------------
    {
      const int r = l16;            // tile row = seq*4 + t
      const int s = r >> 2, t = r & 3;
      const float* hrow = histlds + (nb0 + s) * 16 + t * 4;
      const float h0 = hrow[0], h1 = hrow[1], h2 = hrow[2], h3 = hrow[3];
#pragma unroll
      for (int cc = 0; cc < 32; ++cc) {
        const int c = kh * 32 + cc;
        X[r * 64 + c] = P.in_b[c] + P.pe[t * 64 + c]
                      + h0 * P.in_w[c * 4 + 0] + h1 * P.in_w[c * 4 + 1]
                      + h2 * P.in_w[c * 4 + 2] + h3 * P.in_w[c * 4 + 3];
      }
    }

#pragma unroll 1
    for (int l = 0; l < 2; ++l) {
      const _Float16* Wqkv = P.w16 + l * 32768;
      const _Float16* Wout = Wqkv + 12288;
      const _Float16* Wff1 = Wqkv + 16384;
      const _Float16* Wff2 = Wqkv + 24576;

      // ---- QKV: x(16x64) @ Wqkv.T -> S[16][192] (f16) ---------------------
      {
        const v16h a0 = load_a_f32(X, l16, 64, kh, 0);
        const v16h a1 = load_a_f32(X, l16, 64, kh, 32);
#pragma unroll 1
        for (int nt = 0; nt < 12; ++nt) {
          const int n0 = nt * 16;
          v8f c = {};
          c = wmma16(a0, load_b_f16(Wqkv, n0 + l16, 64, kh, 0), c);
          c = wmma16(a1, load_b_f16(Wqkv, n0 + l16, 64, kh, 32), c);
          const float bias = P.qkv_b[l][n0 + l16];
#pragma unroll
          for (int v = 0; v < 8; ++v)
            S[(v + 8 * kh) * 192 + n0 + l16] = (_Float16)(c[v] + bias);
        }
      }

      // ---- attention (block-diagonal: 4 seqs x 4 heads x T=4) -------------
      // lane = seq*8 + head*2 + tt ; each lane does rows t = 2*tt{+0,1}.
      // Writes o over the q region of S (disjoint per lane; in-wave LDS order).
      {
        const int s  = lane >> 3;
        const int hh = (lane >> 1) & 3;
        const int tt = lane & 1;
#pragma unroll
        for (int t2 = 0; t2 < 2; ++t2) {
          const int row = s * 4 + tt * 2 + t2;
          float qr[16];
#pragma unroll
          for (int d = 0; d < 16; ++d)
            qr[d] = (float)S[row * 192 + hh * 16 + d];
          float sc[4];
#pragma unroll
          for (int sp = 0; sp < 4; ++sp) {
            const int rk = s * 4 + sp;
            float acc = 0.f;
#pragma unroll
            for (int d = 0; d < 16; ++d)
              acc += qr[d] * (float)S[rk * 192 + 64 + hh * 16 + d];
            sc[sp] = acc * 0.25f;  // 1/sqrt(dh=16)
          }
          const float mx = fmaxf(fmaxf(sc[0], sc[1]), fmaxf(sc[2], sc[3]));
          float ex[4], den = 0.f;
#pragma unroll
          for (int sp = 0; sp < 4; ++sp) {
            ex[sp] = __expf(sc[sp] - mx);
            den += ex[sp];
          }
          const float inv = 1.f / den;
          float o[16];
#pragma unroll
          for (int d = 0; d < 16; ++d) o[d] = 0.f;
#pragma unroll
          for (int sp = 0; sp < 4; ++sp) {
            const float w = ex[sp] * inv;
#pragma unroll
            for (int d = 0; d < 16; ++d)
              o[d] += w * (float)S[(s * 4 + sp) * 192 + 128 + hh * 16 + d];
          }
#pragma unroll
          for (int d = 0; d < 16; ++d)
            S[row * 192 + hh * 16 + d] = (_Float16)o[d];
        }
      }

      // ---- out-proj + residual + LN1 --------------------------------------
      {
        const v16h a0 = load_a_f16(S, l16, 192, kh, 0);
        const v16h a1 = load_a_f16(S, l16, 192, kh, 32);
#pragma unroll 1
        for (int nt = 0; nt < 4; ++nt) {
          const int n0 = nt * 16;
          v8f c = {};
          c = wmma16(a0, load_b_f16(Wout, n0 + l16, 64, kh, 0), c);
          c = wmma16(a1, load_b_f16(Wout, n0 + l16, 64, kh, 32), c);
          const float bias = P.out_b[l][n0 + l16];
#pragma unroll
          for (int v = 0; v < 8; ++v)
            X[(v + 8 * kh) * 64 + n0 + l16] += c[v] + bias;
        }
        layernorm_rowhalf(X, l16, kh, P.ln1_g[l], P.ln1_b[l]);
      }

      // ---- FF: gelu(x@W1.T+b1)@W2.T + b2, residual, LN2 -------------------
      {
        const v16h a0 = load_a_f32(X, l16, 64, kh, 0);
        const v16h a1 = load_a_f32(X, l16, 64, kh, 32);
#pragma unroll 1
        for (int nt = 0; nt < 8; ++nt) {
          const int n0 = nt * 16;
          v8f c = {};
          c = wmma16(a0, load_b_f16(Wff1, n0 + l16, 64, kh, 0), c);
          c = wmma16(a1, load_b_f16(Wff1, n0 + l16, 64, kh, 32), c);
          const float bias = P.ff1_b[l][n0 + l16];
#pragma unroll
          for (int v = 0; v < 8; ++v) {
            const float x = c[v] + bias;  // exact GELU (erf)
            S[(v + 8 * kh) * 192 + n0 + l16] =
                (_Float16)(0.5f * x * (1.f + erff(x * 0.70710678118f)));
          }
        }
        const v16h f0 = load_a_f16(S, l16, 192, kh, 0);
        const v16h f1 = load_a_f16(S, l16, 192, kh, 32);
        const v16h f2 = load_a_f16(S, l16, 192, kh, 64);
        const v16h f3 = load_a_f16(S, l16, 192, kh, 96);
#pragma unroll 1
        for (int nt = 0; nt < 4; ++nt) {
          const int n0 = nt * 16;
          v8f c = {};
          c = wmma16(f0, load_b_f16(Wff2, n0 + l16, 128, kh, 0), c);
          c = wmma16(f1, load_b_f16(Wff2, n0 + l16, 128, kh, 32), c);
          c = wmma16(f2, load_b_f16(Wff2, n0 + l16, 128, kh, 64), c);
          c = wmma16(f3, load_b_f16(Wff2, n0 + l16, 128, kh, 96), c);
          const float bias = P.ff2_b[l][n0 + l16];
#pragma unroll
          for (int v = 0; v < 8; ++v)
            X[(v + 8 * kh) * 64 + n0 + l16] += c[v] + bias;
        }
        layernorm_rowhalf(X, l16, kh, P.ln2_g[l], P.ln2_b[l]);
      }
    }  // layers

    // ---- temporal mean over T + type embedding -> nbembed [n][96] ---------
    {
      const int s  = lane >> 3;          // seq in tile
      const int c0 = (lane & 7) * 8;     // 8 columns per lane
      const int nb = nb0 + s;
#pragma unroll
      for (int cc = 0; cc < 8; ++cc) {
        const int c = c0 + cc;
        const float m = 0.25f * (X[(s * 4 + 0) * 64 + c] + X[(s * 4 + 1) * 64 + c] +
                                 X[(s * 4 + 2) * 64 + c] + X[(s * 4 + 3) * 64 + c]);
        nbembed[nb * 96 + c] = (_Float16)m;
      }
      if (lane < 4) {
        const int nb2 = nb0 + lane;
        const int ty  = P.nb_types[b * 32 + nb2];
#pragma unroll
        for (int j = 0; j < 8; ++j)
          nbembed[nb2 * 96 + 64 + j] = (_Float16)P.type_emb[ty * 8 + j];
#pragma unroll
        for (int j = 72; j < 96; ++j) nbembed[nb2 * 96 + j] = (_Float16)0.0f;
      }
    }
  }  // tile loop
  __syncthreads();

  // ============================ GAT social pooling =========================
  // node proj: nbembed(32x96,f16) @ node_w.T(96->64) -> hbuf (8 WMMA tiles)
  {
    const _Float16* Wn = P.w16 + 65536;  // [64][96] padded
#pragma unroll 1
    for (int i = 0; i < 2; ++i) {
      const int tileid = wv * 2 + i;
      const int m0 = (tileid >> 2) * 16, n0 = (tileid & 3) * 16;
      v8f c = {};
#pragma unroll
      for (int kt = 0; kt < 3; ++kt) {
        const v16h a  = load_a_f16(nbembed, m0 + l16, 96, kh, kt * 32);
        const v16h bb = load_b_f16(Wn, n0 + l16, 96, kh, kt * 32);
        c = wmma16(a, bb, c);
      }
#pragma unroll
      for (int v = 0; v < 8; ++v)
        hbuf[(m0 + v + 8 * kh) * 64 + n0 + l16] = c[v];
    }
  }
  __syncthreads();

  // attention logits: 128 threads = 32 n x 4 heads
  {
    const int n = tid >> 2, hh = tid & 3;
    float acc = 0.f;
#pragma unroll
    for (int d = 0; d < 16; ++d)
      acc += hbuf[n * 64 + hh * 16 + d] * (P.a_src[d] + P.a_dst[d]);
    const float* th = P.target_hist + (size_t)b * 16;  // row T-1 = 3
    const float tpx = th[12], tpy = th[13], tvx = th[14], tvy = th[15];
    const float* nh = histlds + n * 16 + 12;           // staged last timestep
    const float dx = nh[0] - tpx, dy = nh[1] - tpy;
    const float dist = sqrtf(dx * dx + dy * dy);
    const float rx = nh[2] - tvx, ry = nh[3] - tvy;
    const float closing = -(dx * rx + dy * ry) / (dist + 1e-8f);
    const float ttc = fminf(fmaxf(dist / (fabsf(closing) + 1e-6f), 0.f), 10.f);
    acc += dist * P.edge_w[hh * 4 + 0] + rx * P.edge_w[hh * 4 + 1] +
           ry * P.edge_w[hh * 4 + 2] + ttc * P.edge_w[hh * 4 + 3];
    acc = (acc > 0.f) ? acc : 0.2f * acc;  // leaky relu
    logitb[n * 4 + hh] = P.nb_mask[b * 32 + n] ? acc : -1e30f;
  }
  __syncthreads();

  // softmax over n per head (4 heads)
  if (tid < 4) {
    const int hh = tid;
    float mx = -1e30f;
    for (int n = 0; n < 32; ++n) mx = fmaxf(mx, logitb[n * 4 + hh]);
    float ex[32];
    float den = 0.f;
    for (int n = 0; n < 32; ++n) {
      const float lg = logitb[n * 4 + hh];
      const float e  = (lg > -1e29f) ? __expf(lg - mx) : 0.f;
      ex[n] = e;
      den += e;
    }
    const float inv = (den > 0.f) ? 1.f / den : 0.f;
    for (int n = 0; n < 32; ++n) wattn[n * 4 + hh] = ex[n] * inv;
  }
  __syncthreads();

  // weighted sum over neighbors: 64 threads = 4 heads x 16 dims
  if (tid < 64) {
    const int hh = tid >> 4, d = tid & 15;
    float acc = 0.f;
    for (int n = 0; n < 32; ++n)
      acc += wattn[n * 4 + hh] * hbuf[n * 64 + hh * 16 + d];
    wsum[hh * 16 + d] = acc;
  }
  __syncthreads();

  // output projection: social = weighted @ gout_w.T + gout_b, gated by has_nb
  if (tid < 64) {
    int any = 0;
    for (int n = 0; n < 32; ++n) any |= P.nb_mask[b * 32 + n];
    float acc = P.gout_b[tid];
    for (int d = 0; d < 64; ++d) acc += P.gout_w[tid * 64 + d] * wsum[d];
    P.out[(size_t)b * 64 + tid] = any ? acc : 0.f;
  }
}

// ---------------------------------------------------------------------------
// Host launcher. Input flattening assumption (setup_inputs dict insertion
// order, params dict/list recursively in insertion order):
//  0 target_embed, 1 target_hist, 2 nb_hists, 3 nb_types, 4 nb_mask,
//  5 in_w, 6 in_b, 7 pe, 8 type_emb,
//  9+12*l: qkv_w qkv_b out_w out_b ln1_g ln1_b ff1_w ff1_b ff2_w ff2_b ln2_g ln2_b
//  33 node_w, 34 edge_w, 35 a_src, 36 a_dst, 37 gout_w, 38 gout_b
// ---------------------------------------------------------------------------
extern "C" void kernel_launch(void* const* d_in, const int* in_sizes, int n_in,
                              void* d_out, int out_size, void* d_ws,
                              size_t ws_size, hipStream_t stream) {
  (void)n_in; (void)out_size; (void)ws_size;
  const float* target_hist = (const float*)d_in[1];
  const float* nb_hists    = (const float*)d_in[2];
  const int*   nb_types    = (const int*)d_in[3];
  const int*   nb_mask     = (const int*)d_in[4];

  _Float16* w16 = (_Float16*)d_ws;  // 71680 halves = 143360 bytes

  PrepArgs pa;
  MainArgs ma;
  ma.nb_hists    = nb_hists;
  ma.nb_types    = nb_types;
  ma.nb_mask     = nb_mask;
  ma.target_hist = target_hist;
  ma.in_w     = (const float*)d_in[5];
  ma.in_b     = (const float*)d_in[6];
  ma.pe       = (const float*)d_in[7];
  ma.type_emb = (const float*)d_in[8];
  for (int l = 0; l < 2; ++l) {
    const int base = 9 + 12 * l;
    pa.qkv_w[l] = (const float*)d_in[base + 0];
    ma.qkv_b[l] = (const float*)d_in[base + 1];
    pa.out_w[l] = (const float*)d_in[base + 2];
    ma.out_b[l] = (const float*)d_in[base + 3];
    ma.ln1_g[l] = (const float*)d_in[base + 4];
    ma.ln1_b[l] = (const float*)d_in[base + 5];
    pa.ff1_w[l] = (const float*)d_in[base + 6];
    ma.ff1_b[l] = (const float*)d_in[base + 7];
    pa.ff2_w[l] = (const float*)d_in[base + 8];
    ma.ff2_b[l] = (const float*)d_in[base + 9];
    ma.ln2_g[l] = (const float*)d_in[base + 10];
    ma.ln2_b[l] = (const float*)d_in[base + 11];
  }
  pa.node_w = (const float*)d_in[33];
  ma.edge_w = (const float*)d_in[34];
  ma.a_src  = (const float*)d_in[35];
  ma.a_dst  = (const float*)d_in[36];
  ma.gout_w = (const float*)d_in[37];
  ma.gout_b = (const float*)d_in[38];
  ma.w16 = w16;
  ma.out = (float*)d_out;

  hipLaunchKernelGGL(prep_weights_f16, dim3(128), dim3(256), 0, stream, pa, w16);

  const int B = in_sizes[2] / (32 * 4 * 4);  // batch from nb_hists size
  hipLaunchKernelGGL(gat_social_fused, dim3(B), dim3(128), 0, stream, ma);
}
